// MyLSTM_82652350644575
// MI455X (gfx1250) — compile-verified, matrix operand order checked by reading, not run on hardware
//
#include <hip/hip_runtime.h>
#include <hip/hip_bf16.h>
#include <math.h>

typedef __bf16 bf16_t;
typedef __attribute__((ext_vector_type(16))) __bf16 v16bf;
typedef __attribute__((ext_vector_type(8)))  __bf16 v8bf;
typedef __attribute__((ext_vector_type(8)))  float  v8f;

#define UNITS  1024
#define NGATE  4096     // 4*UNITS
#define KDIM   1024     // INPUT_SIZE
#define TSTEPS 512
#define BATCH  64
#define NBLK   64       // blocks in the persistent recurrence kernel
#define RT_PAD 8        // bf16 elements of per-row LDS padding (16B -> 4-bank skew)

// Load one 16x32 bf16 WMMA fragment chunk pair.
// base = &M[row][k0 + (lane>>4)*8]; grabs [kk..kk+8) and [kk+16..kk+24).
__device__ __forceinline__ v16bf ld_frag_row(const bf16_t* __restrict__ base) {
  v8bf lo = *(const v8bf*)(base);
  v8bf hi = *(const v8bf*)(base + 16);
  v16bf r;
#pragma unroll
  for (int i = 0; i < 8; ++i) { r[i] = lo[i]; r[8 + i] = hi[i]; }
  return r;
}

// src[K][N] (f32) -> dst[N][K] (bf16); writes coalesced along K.
__global__ void k_transpose_bf16(const float* __restrict__ src,
                                 bf16_t* __restrict__ dst, int K, int N) {
  long i = (long)blockIdx.x * blockDim.x + threadIdx.x;
  long total = (long)K * N;
  if (i >= total) return;
  int k = (int)(i % K);
  int n = (int)(i / K);
  dst[i] = (bf16_t)src[(long)k * N + n];
}

__global__ void k_f32_to_bf16(const float* __restrict__ src,
                              bf16_t* __restrict__ dst, long n) {
  long i = (long)blockIdx.x * blockDim.x + threadIdx.x;
  if (i < n) dst[i] = (bf16_t)src[i];
}

__global__ void k_init_state(bf16_t* __restrict__ h0, float* __restrict__ c,
                             unsigned* __restrict__ bar) {
  int i = blockIdx.x * blockDim.x + threadIdx.x;   // BATCH*UNITS threads
  h0[i] = (bf16_t)0.0f;
  c[i]  = 0.0f;
  if (i == 0) *bar = 0u;
}

// XZ[m][n] = bias[n] + sum_k X[m][k] * Wt[n][k]
// M=32768, N=4096, K=1024. Each wave: 32(M) x 64(N) tile, bf16 WMMA, f32 accum.
__global__ void __launch_bounds__(256) k_input_gemm(
    const bf16_t* __restrict__ X,    // [32768][1024] bf16
    const bf16_t* __restrict__ Wt,   // [4096][1024]  bf16 (transposed)
    const float*  __restrict__ bias, // [4096]
    float* __restrict__ XZ)          // [32768][4096] f32
{
  int lane  = threadIdx.x & 31;
  int wid   = blockIdx.x * 8 + (threadIdx.x >> 5);
  int m0    = (wid >> 6) * 32;     // 1024 m-blocks
  int n0    = (wid & 63) * 64;     // 64 n-blocks
  int lrow  = lane & 15;
  int lhalf = lane >> 4;

  v8f acc[2][4];
#pragma unroll
  for (int nt = 0; nt < 4; ++nt) {
    float b = bias[n0 + nt * 16 + lrow];
#pragma unroll
    for (int mt = 0; mt < 2; ++mt)
#pragma unroll
      for (int r = 0; r < 8; ++r) acc[mt][nt][r] = b;
  }

  for (int k0 = 0; k0 < KDIM; k0 += 32) {
    int kk = k0 + lhalf * 8;
    v16bf a[2], b[4];
#pragma unroll
    for (int mt = 0; mt < 2; ++mt)
      a[mt] = ld_frag_row(X + (long)(m0 + mt * 16 + lrow) * KDIM + kk);
#pragma unroll
    for (int nt = 0; nt < 4; ++nt)
      b[nt] = ld_frag_row(Wt + (long)(n0 + nt * 16 + lrow) * KDIM + kk);
#pragma unroll
    for (int mt = 0; mt < 2; ++mt)
#pragma unroll
      for (int nt = 0; nt < 4; ++nt)
        acc[mt][nt] = __builtin_amdgcn_wmma_f32_16x16x32_bf16(
            false, a[mt], false, b[nt], (short)0, acc[mt][nt], false, false);
  }

#pragma unroll
  for (int mt = 0; mt < 2; ++mt)
#pragma unroll
    for (int nt = 0; nt < 4; ++nt)
#pragma unroll
      for (int r = 0; r < 8; ++r) {
        int row = m0 + mt * 16 + r + lhalf * 8;
        int col = n0 + nt * 16 + lrow;
        XZ[(long)row * NGATE + col] = acc[mt][nt][r];
      }
}

// Persistent recurrence kernel: 64 blocks x 128 threads (4 waves = 4 gates).
// Each block owns a 16-wide U slice; its 64 Rt rows live in LDS for all 512
// steps. Per step: WMMA z = xz_t + h@R (B from LDS), gate fusion through LDS,
// then a device-scope grid barrier before the next step.
__global__ void __launch_bounds__(128) k_lstm_persistent(
    const float*  __restrict__ XZ,    // [512*64][4096] precomputed x@W + bias
    const bf16_t* __restrict__ Rt,    // [4096][1024] bf16 (transposed)
    bf16_t* __restrict__ hA,          // [64][1024] bf16 (zero-initialized)
    bf16_t* __restrict__ hB,          // [64][1024] bf16
    float*  __restrict__ c,           // [64][1024] f32 (zero-initialized)
    float*  __restrict__ out,         // [512][64][1024] f32
    unsigned* __restrict__ bar)       // grid-barrier arrive counter (init 0)
{
  __shared__ __align__(16) bf16_t rts[NBLK][KDIM + RT_PAD]; // 64 rows, 2064B stride
  __shared__ float zs[4][BATCH][16];

  int lane  = threadIdx.x & 31;
  int g     = threadIdx.x >> 5;       // gate: 0=i 1=f 2=c 3=o
  int u0    = blockIdx.x * 16;
  int n0    = g * UNITS + u0;         // column slice in the 4096-wide z
  int lrow  = lane & 15;
  int lhalf = lane >> 4;

  // ---- Stage this block's 64 Rt rows (wave g: rows n0..n0+15) into LDS ----
#pragma unroll 1
  for (int r = 0; r < 16; ++r) {
    const bf16_t* src = Rt + (long)(n0 + r) * KDIM;
    bf16_t* dst = &rts[g * 16 + r][0];
    for (int ch = lane; ch < KDIM / 8; ch += 32)       // 16B chunks
      *(v8bf*)(dst + ch * 8) = *(const v8bf*)(src + ch * 8);
  }
  __syncthreads();

  const int lrbase = g * 16 + lrow;   // local LDS row for the B fragment

  for (int t = 0; t < TSTEPS; ++t) {
    const bf16_t* h_in  = (t & 1) ? hB : hA;
    bf16_t*       h_out = (t & 1) ? hA : hB;
    const float*  xz_t  = XZ + (long)t * BATCH * NGATE;
    float*        out_t = out + (long)t * BATCH * UNITS;

    // C init = xz_t tile (f32 C/D layout: vgpr r -> M = r + lhalf*8, N = lrow)
    v8f acc[4];
#pragma unroll
    for (int mt = 0; mt < 4; ++mt)
#pragma unroll
      for (int r = 0; r < 8; ++r)
        acc[mt][r] = xz_t[(long)(mt * 16 + r + lhalf * 8) * NGATE + (n0 + lrow)];

    for (int k0 = 0; k0 < UNITS; k0 += 32) {
      int kk = k0 + lhalf * 8;
      // B fragment from LDS (padded rows -> conflict-skewed banks)
      const bf16_t* bb = &rts[lrbase][kk];
      v8bf blo = *(const v8bf*)(bb);
      v8bf bhi = *(const v8bf*)(bb + 16);
      v16bf bfrag;
#pragma unroll
      for (int i = 0; i < 8; ++i) { bfrag[i] = blo[i]; bfrag[8 + i] = bhi[i]; }
      // A fragments from global h (L2-hot, 128 KB)
      v16bf a[4];
#pragma unroll
      for (int mt = 0; mt < 4; ++mt)
        a[mt] = ld_frag_row(h_in + (long)(mt * 16 + lrow) * UNITS + kk);
#pragma unroll
      for (int mt = 0; mt < 4; ++mt)
        acc[mt] = __builtin_amdgcn_wmma_f32_16x16x32_bf16(
            false, a[mt], false, bfrag, (short)0, acc[mt], false, false);
    }

#pragma unroll
    for (int mt = 0; mt < 4; ++mt)
#pragma unroll
      for (int r = 0; r < 8; ++r)
        zs[g][mt * 16 + r + lhalf * 8][lrow] = acc[mt][r];

    __syncthreads();

    // Gate fusion: 64x16 = 1024 cells, 8 per thread.
#pragma unroll
    for (int i = 0; i < 8; ++i) {
      int e = threadIdx.x * 8 + i;
      int m = e >> 4;
      int j = e & 15;
      float zi = zs[0][m][j];
      float zf = zs[1][m][j];
      float zc = zs[2][m][j];
      float zo = zs[3][m][j];
      float ig = 1.0f / (1.0f + __expf(-zi));
      float fg = 1.0f / (1.0f + __expf(-zf));
      float gg = tanhf(zc);
      float og = 1.0f / (1.0f + __expf(-zo));
      long idx = (long)m * UNITS + (u0 + j);
      float cn = fg * c[idx] + ig * gg;
      float hn = og * tanhf(cn);
      c[idx]     = cn;
      out_t[idx] = hn;
      h_out[idx] = (bf16_t)hn;
    }

    // ---- Grid barrier: make h_out visible device-wide, then rendezvous ----
    __syncthreads();          // zs consumed; block's stores all issued
    __threadfence();          // flush this thread's stores to device scope
    if (threadIdx.x == 0) {
      __hip_atomic_fetch_add(bar, 1u, __ATOMIC_RELEASE, __HIP_MEMORY_SCOPE_AGENT);
      unsigned target = (unsigned)NBLK * (unsigned)(t + 1);
      while (__hip_atomic_load(bar, __ATOMIC_ACQUIRE, __HIP_MEMORY_SCOPE_AGENT) < target)
        __builtin_amdgcn_s_sleep(1);
    }
    __syncthreads();
    __threadfence();          // acquire side for all threads
  }
}

__global__ void k_finalize(const float* __restrict__ h_last,
                           const float* __restrict__ c,
                           float* __restrict__ ht_out,
                           float* __restrict__ ct_out) {
  int i = blockIdx.x * blockDim.x + threadIdx.x;  // BATCH*UNITS threads
  ht_out[i] = h_last[i];
  ct_out[i] = c[i];
}

extern "C" void kernel_launch(void* const* d_in, const int* in_sizes, int n_in,
                              void* d_out, int out_size, void* d_ws, size_t ws_size,
                              hipStream_t stream) {
  (void)in_sizes; (void)n_in; (void)out_size; (void)ws_size;

  const float* inputs = (const float*)d_in[0];  // [512,64,1024]
  const float* Wk     = (const float*)d_in[1];  // [1024,4096]
  const float* Rk     = (const float*)d_in[2];  // [1024,4096]
  const float* bias   = (const float*)d_in[3];  // [4096]
  float* out = (float*)d_out;

  char* ws = (char*)d_ws;
  size_t off = 0;
  auto take = [&](size_t bytes) -> char* {
    char* p = ws + off;
    off += (bytes + 255) & ~(size_t)255;
    return p;
  };

  bf16_t*   Wt  = (bf16_t*)take((size_t)NGATE * KDIM * 2);          // 8 MB
  bf16_t*   Rt  = (bf16_t*)take((size_t)NGATE * UNITS * 2);         // 8 MB
  bf16_t*   Xb  = (bf16_t*)take((size_t)TSTEPS * BATCH * KDIM * 2); // 64 MB
  float*    XZ  = (float*)take((size_t)TSTEPS * BATCH * NGATE * 4); // 512 MB
  bf16_t*   h0  = (bf16_t*)take((size_t)BATCH * UNITS * 2);
  bf16_t*   h1  = (bf16_t*)take((size_t)BATCH * UNITS * 2);
  float*    cst = (float*)take((size_t)BATCH * UNITS * 4);
  unsigned* bar = (unsigned*)take(256);

  // 1) Weight transpose + bf16 convert (both [K][N] -> [N][K]).
  {
    long tot = (long)KDIM * NGATE;
    unsigned nb = (unsigned)((tot + 255) / 256);
    k_transpose_bf16<<<nb, 256, 0, stream>>>(Wk, Wt, KDIM, NGATE);
    k_transpose_bf16<<<nb, 256, 0, stream>>>(Rk, Rt, UNITS, NGATE);
  }
  // 2) Inputs f32 -> bf16.
  {
    long tot = (long)TSTEPS * BATCH * KDIM;
    k_f32_to_bf16<<<(unsigned)((tot + 255) / 256), 256, 0, stream>>>(inputs, Xb, tot);
  }
  // 3) Big parallel input GEMM: XZ = X@W + bias.
  k_input_gemm<<<8192, 256, 0, stream>>>(Xb, Wt, bias, XZ);
  // 4) Zero initial state + barrier counter.
  k_init_state<<<(BATCH * UNITS) / 256, 256, 0, stream>>>(h0, cst, bar);
  // 5) Entire recurrence in ONE persistent kernel (grid barrier per step).
  k_lstm_persistent<<<NBLK, 128, 0, stream>>>(XZ, Rt, h0, h1, cst, out, bar);
  // 6) Append final (ht, ct) after the [T,B,U] outputs.
  k_finalize<<<(BATCH * UNITS) / 256, 256, 0, stream>>>(
      out + (size_t)(TSTEPS - 1) * BATCH * UNITS, cst,
      out + (size_t)TSTEPS * BATCH * UNITS,
      out + (size_t)TSTEPS * BATCH * UNITS + BATCH * UNITS);
}
